// MultiHeadAttention_47072841564835
// MI455X (gfx1250) — matrix operand order repaired and checked by reading, not compiled, not run on hardware
//
#include <hip/hip_runtime.h>
#include <hip/hip_bf16.h>

#define BB  2
#define SS  2048
#define DD  1024
#define HH  16
#define DKV 64

typedef __attribute__((ext_vector_type(16))) __bf16 v16bf;
typedef __attribute__((ext_vector_type(8)))  float  v8f;
typedef __attribute__((ext_vector_type(4)))  unsigned int u32x4;
typedef __attribute__((ext_vector_type(8)))  int  i32x8;
typedef __attribute__((ext_vector_type(4)))  int  i32x4;

#if __has_builtin(__builtin_amdgcn_tensor_load_to_lds) && \
    __has_builtin(__builtin_amdgcn_s_wait_tensorcnt)
#define HAVE_TDM 1
#else
#define HAVE_TDM 0
#endif

union FragBF { uint4 u4[2]; v16bf v; };
union U8x16  { uint4 u4; unsigned short s[8]; };

__device__ __forceinline__ unsigned short f2bf(float f) {
  unsigned int u = __float_as_uint(f);
  u += 0x7fffu + ((u >> 16) & 1u);           // round-to-nearest-even
  return (unsigned short)(u >> 16);
}
__device__ __forceinline__ unsigned int pack2bf(float a, float b) {
  return (unsigned int)f2bf(a) | ((unsigned int)f2bf(b) << 16);
}
__device__ __forceinline__ v8f wmma_bf16(const FragBF& a, const FragBF& b, v8f c) {
  return __builtin_amdgcn_wmma_f32_16x16x32_bf16(false, a.v, false, b.v,
                                                 (short)0, c, false, false);
}

// ---------------------------------------------------------------------------
// C[M,N] = A[M,K] @ W[N,K]^T   (A fp32 or bf16 via template, W fp32)
// MODE 0: write bf16 head-major [B,H,S,64]   MODE 1: write fp32 row-major
// Block: 256 threads = 8 waves (4 m-rows x 2 n-cols), tile 64x64, BK=32.
// When A is bf16 and the TDM builtins exist, the A-tile (64x32 bf16, padded
// LDS rows of 80B) is staged with tensor_load_to_lds issued by wave 0.
// ---------------------------------------------------------------------------
template<int A_BF16, int MODE>
__global__ __launch_bounds__(256)
void gemm_xwt_kernel(const void* Aptr, const float* W,
                     int M, int N, int K,
                     unsigned short* outBf, float* outF)
{
  // single static LDS block: As at byte offset 0 (assumed for the D#),
  // Bs at byte offset 5120. Rows padded to 20 dwords (80B, 16B-aligned).
  __shared__ __align__(16) unsigned int SMEM[2 * 64 * 20];
  unsigned int* As = SMEM;
  unsigned int* Bs = SMEM + 64 * 20;

  const int tid  = threadIdx.x, lane = tid & 31, wave = tid >> 5;
  const int wm   = wave >> 1,   wn   = wave & 1;
  const int half = lane >> 4,   mr   = lane & 15;
  const int m0   = blockIdx.y * 64, n0 = blockIdx.x * 64;

  v8f acc0 = {0.f,0.f,0.f,0.f,0.f,0.f,0.f,0.f};
  v8f acc1 = acc0;

  const int srow = tid >> 2;        // 0..63
  const int scol = (tid & 3) * 8;   // 0,8,16,24 (elements)

  for (int k0 = 0; k0 < K; k0 += 32) {
    // ---- stage A tile ----
    if constexpr (A_BF16) {
#if HAVE_TDM
      if (wave == 0) {
        // D# for a 2D bf16 tensor [M x K], tile 64x32, dest LDS offset 0,
        // pad 4 dwords after every 16 dwords (row stride 80B).
        const unsigned long long ga =
            (unsigned long long)(const unsigned short*)Aptr +
            ((unsigned long long)m0 * (unsigned)K + (unsigned)k0) * 2ull;
        u32x4 g0 = { 1u,                                   // count=1, no gather
                     0u,                                   // lds_addr = 0
                     (unsigned int)ga,
                     (unsigned int)((ga >> 32) & 0x1FFFFFFu) | (2u << 30) }; // type=2
        const unsigned int ud0 = (1u << 16)   // data_size = 2B
                               | (1u << 20)   // pad_enable
                               | (3u << 22)   // pad_interval: 16 dwords
                               | (3u << 25);  // pad_amount: 4 dwords
        i32x8 g1 = { (int)ud0,
                     (int)(((unsigned)K & 0xFFFFu) << 16),                    // tensor_dim0 lo
                     (int)((((unsigned)K >> 16) & 0xFFFFu) |
                           (((unsigned)M & 0xFFFFu) << 16)),                  // dim0 hi | dim1 lo
                     (int)((((unsigned)M >> 16) & 0xFFFFu) | (32u << 16)),    // dim1 hi | tile_dim0=32
                     (int)64,                                                 // tile_dim1=64
                     (int)(unsigned)K,                                        // tensor_dim0_stride lo
                     0, 0 };
        i32x4 gz = {0, 0, 0, 0};
#if __clang_major__ >= 23
        i32x8 gz8 = {0,0,0,0,0,0,0,0};
        __builtin_amdgcn_tensor_load_to_lds(g0, g1, gz, gz, gz8, 0);
#else
        __builtin_amdgcn_tensor_load_to_lds(g0, g1, gz, gz, 0);
#endif
        __builtin_amdgcn_s_wait_tensorcnt(0);
      }
#else
      const unsigned short* src = (const unsigned short*)Aptr
                                  + (size_t)(m0 + srow) * K + k0 + scol;
      uint4 x = *(const uint4*)src;
      unsigned int* dst = &As[srow * 20 + scol / 2];
      dst[0] = x.x; dst[1] = x.y; dst[2] = x.z; dst[3] = x.w;
#endif
    } else {
      const float* src = (const float*)Aptr + (size_t)(m0 + srow) * K + k0 + scol;
      float4 x = *(const float4*)src;
      float4 y = *(const float4*)(src + 4);
      unsigned int* dst = &As[srow * 20 + scol / 2];
      dst[0] = pack2bf(x.x, x.y); dst[1] = pack2bf(x.z, x.w);
      dst[2] = pack2bf(y.x, y.y); dst[3] = pack2bf(y.z, y.w);
    }
    // ---- stage B tile (fp32 weights -> bf16) ----
    {
      const float* src = W + (size_t)(n0 + srow) * K + k0 + scol;
      float4 x = *(const float4*)src;
      float4 y = *(const float4*)(src + 4);
      unsigned int* dst = &Bs[srow * 20 + scol / 2];
      dst[0] = pack2bf(x.x, x.y); dst[1] = pack2bf(x.z, x.w);
      dst[2] = pack2bf(y.x, y.y); dst[3] = pack2bf(y.z, y.w);
    }
    __syncthreads();

    FragBF a;
    const unsigned int* ar = &As[(wm * 16 + mr) * 20];
    a.u4[0] = *(const uint4*)(ar + half * 4);        // K = half*8 .. +7
    a.u4[1] = *(const uint4*)(ar + 8 + half * 4);    // K = 16+half*8 .. +7

    FragBF b0, b1;
    const unsigned int* br0 = &Bs[(wn * 32 + mr) * 20];
    b0.u4[0] = *(const uint4*)(br0 + half * 4);      // K = half*8 .. +7
    b0.u4[1] = *(const uint4*)(br0 + 8 + half * 4);  // K = 16+half*8 .. +7
    const unsigned int* br1 = &Bs[(wn * 32 + 16 + mr) * 20];
    b1.u4[0] = *(const uint4*)(br1 + half * 4);
    b1.u4[1] = *(const uint4*)(br1 + 8 + half * 4);

    acc0 = wmma_bf16(a, b0, acc0);
    acc1 = wmma_bf16(a, b1, acc1);
    __syncthreads();
  }

  #pragma unroll
  for (int r = 0; r < 8; r++) {
    const int m   = m0 + wm * 16 + r + half * 8;
    const int n_a = n0 + wn * 32 + mr;
    const int n_b = n_a + 16;
    const float va = acc0[r], vb = acc1[r];
    if constexpr (MODE == 0) {
      const int bb = m >> 11, s = m & 2047;
      { int h = n_a >> 6, d = n_a & 63;
        outBf[(((size_t)(bb * HH + h)) * SS + s) * DKV + d] = f2bf(va); }
      { int h = n_b >> 6, d = n_b & 63;
        outBf[(((size_t)(bb * HH + h)) * SS + s) * DKV + d] = f2bf(vb); }
    } else {
      outF[(size_t)m * N + n_a] = va;
      outF[(size_t)m * N + n_b] = vb;
    }
  }
}

// ---------------------------------------------------------------------------
// Scores + mask + softmax.  Block = 256 thr (8 waves) handles 16 queries x
// all 2048 keys.  LDS holds the fp32 score row block (16 x 2056 = 128.5 KB).
// ---------------------------------------------------------------------------
__global__ __launch_bounds__(256)
void attn_scores_softmax_kernel(const unsigned short* Qb, const unsigned short* Kb,
                                const unsigned char* mask, float* attn)
{
  extern __shared__ float sc[];            // [16][2056]
  const int SROW = 2056;
  const int bh = blockIdx.y, b = bh >> 4;
  const int q0 = blockIdx.x * 16;
  const int tid = threadIdx.x, lane = tid & 31, wave = tid >> 5;
  const int half = lane >> 4, mr = lane & 15;

  // Q fragments for K-dims 0..31 and 32..63 (head-major bf16, 128B rows)
  const unsigned short* qrow = Qb + ((size_t)bh * SS + q0 + mr) * DKV;
  FragBF a0, a1;
  a0.u4[0] = *(const uint4*)(qrow + half * 8);
  a0.u4[1] = *(const uint4*)(qrow + 16 + half * 8);
  a1.u4[0] = *(const uint4*)(qrow + 32 + half * 8);
  a1.u4[1] = *(const uint4*)(qrow + 48 + half * 8);

  for (int i = 0; i < 16; i++) {
    const int key0 = (i * 8 + wave) * 16;
    const unsigned short* krow = Kb + ((size_t)bh * SS + key0 + mr) * DKV;
    FragBF b0, b1;
    b0.u4[0] = *(const uint4*)(krow + half * 8);
    b0.u4[1] = *(const uint4*)(krow + 16 + half * 8);
    b1.u4[0] = *(const uint4*)(krow + 32 + half * 8);
    b1.u4[1] = *(const uint4*)(krow + 48 + half * 8);

    v8f acc = {0.f,0.f,0.f,0.f,0.f,0.f,0.f,0.f};
    acc = wmma_bf16(a0, b0, acc);
    acc = wmma_bf16(a1, b1, acc);

    #pragma unroll
    for (int r = 0; r < 8; r++) {
      const int q = r + half * 8;
      const unsigned char mv = mask[((size_t)b * SS + q0 + q) * SS + key0 + mr];
      sc[q * SROW + key0 + mr] = mv ? -1e9f : acc[r] * 0.125f;
    }
  }
  __syncthreads();

  // Each wave owns 2 rows; lane-strided scan + wave reductions.
  for (int rr = 0; rr < 2; rr++) {
    const int q = wave * 2 + rr;
    float* row = &sc[q * SROW];
    float mx = -3.4e38f;
    for (int j = lane; j < SS; j += 32) mx = fmaxf(mx, row[j]);
    for (int o = 16; o > 0; o >>= 1) mx = fmaxf(mx, __shfl_xor(mx, o));
    float sum = 0.f;
    for (int j = lane; j < SS; j += 32) { float e = __expf(row[j] - mx); row[j] = e; sum += e; }
    for (int o = 16; o > 0; o >>= 1) sum += __shfl_xor(sum, o);
    const float inv = 1.f / sum;
    float* dst = attn + ((size_t)bh * SS + q0 + q) * SS;
    for (int j = lane; j < SS; j += 32) dst[j] = row[j] * inv;
  }
}

// ---------------------------------------------------------------------------
// context[b,s,h*64+d] = sum_k attn[bh,s,k] * V[bh,k,d]
// Block = 128 thr (4 waves, one 16-wide d-tile each) per 16 queries.
// ---------------------------------------------------------------------------
__global__ __launch_bounds__(128)
void attn_context_kernel(const float* attn, const unsigned short* Vb,
                         unsigned short* ctx)
{
  __shared__ __align__(16) unsigned int  Atile[16 * 20];  // attn 16q x 32k bf16
  __shared__ __align__(16) unsigned short Vt[64 * 40];    // V^T [d][key]
  const int bh = blockIdx.y, b = bh >> 4, h = bh & 15;
  const int q0 = blockIdx.x * 16;
  const int tid = threadIdx.x, lane = tid & 31, wave = tid >> 5;
  const int half = lane >> 4, mr = lane & 15;

  v8f acc = {0.f,0.f,0.f,0.f,0.f,0.f,0.f,0.f};
  const int arow = tid >> 3, acol = (tid & 7) * 4;
  const int vkey = tid & 31, vdb = (tid >> 5) * 16;

  for (int k0 = 0; k0 < SS; k0 += 32) {
    {
      const float* src = attn + ((size_t)bh * SS + q0 + arow) * SS + k0 + acol;
      float4 x = *(const float4*)src;
      unsigned int* dst = &Atile[arow * 20 + acol / 2];
      dst[0] = pack2bf(x.x, x.y); dst[1] = pack2bf(x.z, x.w);
    }
    {
      const unsigned short* src = Vb + ((size_t)bh * SS + k0 + vkey) * DKV + vdb;
      U8x16 x, y;
      x.u4 = *(const uint4*)src;
      y.u4 = *(const uint4*)(src + 8);
      #pragma unroll
      for (int j = 0; j < 8; j++) {
        Vt[(vdb + j)     * 40 + vkey] = x.s[j];
        Vt[(vdb + 8 + j) * 40 + vkey] = y.s[j];
      }
    }
    __syncthreads();

    FragBF a;
    const unsigned int* ar = &Atile[mr * 20];
    a.u4[0] = *(const uint4*)(ar + half * 4);
    a.u4[1] = *(const uint4*)(ar + 8 + half * 4);
    FragBF bf;
    const unsigned short* br = &Vt[(wave * 16 + mr) * 40];
    bf.u4[0] = *(const uint4*)(br + half * 8);
    bf.u4[1] = *(const uint4*)(br + 16 + half * 8);
    acc = wmma_bf16(a, bf, acc);
    __syncthreads();
  }

  #pragma unroll
  for (int r = 0; r < 8; r++) {
    const int s   = q0 + r + half * 8;
    const int col = h * DKV + wave * 16 + mr;
    ctx[((size_t)b * SS + s) * (HH * DKV) + col] = f2bf(acc[r]);
  }
}

// ---------------------------------------------------------------------------
// out = LayerNorm(pre + input_Q)  per row of 1024, gamma=1 beta=0 eps=1e-5
// ---------------------------------------------------------------------------
__global__ __launch_bounds__(256)
void ln_residual_kernel(const float* pre, const float* inQ, float* out)
{
  const int row = blockIdx.x, tid = threadIdx.x;
  const int lane = tid & 31, wave = tid >> 5;
  __shared__ float red[8];
  const float* p = pre + (size_t)row * DD;
  const float* q = inQ + (size_t)row * DD;

  float x[4]; float s = 0.f;
  #pragma unroll
  for (int i = 0; i < 4; i++) { x[i] = p[tid + i * 256] + q[tid + i * 256]; s += x[i]; }
  for (int o = 16; o > 0; o >>= 1) s += __shfl_xor(s, o);
  if (lane == 0) red[wave] = s;
  __syncthreads();
  float tot = 0.f;
  #pragma unroll
  for (int i = 0; i < 8; i++) tot += red[i];
  const float mu = tot * (1.f / DD);
  __syncthreads();

  float v = 0.f;
  #pragma unroll
  for (int i = 0; i < 4; i++) { const float d = x[i] - mu; v += d * d; }
  for (int o = 16; o > 0; o >>= 1) v += __shfl_xor(v, o);
  if (lane == 0) red[wave] = v;
  __syncthreads();
  float vt = 0.f;
  #pragma unroll
  for (int i = 0; i < 8; i++) vt += red[i];
  const float inv = rsqrtf(vt * (1.f / DD) + 1e-5f);

  float* o_ = out + (size_t)row * DD;
  #pragma unroll
  for (int i = 0; i < 4; i++) o_[tid + i * 256] = (x[i] - mu) * inv;
}

// ---------------------------------------------------------------------------
extern "C" void kernel_launch(void* const* d_in, const int* in_sizes, int n_in,
                              void* d_out, int out_size, void* d_ws, size_t ws_size,
                              hipStream_t stream)
{
  (void)in_sizes; (void)n_in; (void)out_size; (void)ws_size;
  const float* inQ  = (const float*)d_in[0];
  const float* inK  = (const float*)d_in[1];
  const float* inV  = (const float*)d_in[2];
  const unsigned char* mask = (const unsigned char*)d_in[3];
  const float* Wq   = (const float*)d_in[4];
  const float* Wk   = (const float*)d_in[5];
  const float* Wv   = (const float*)d_in[6];
  const float* Wfc  = (const float*)d_in[7];

  float* out  = (float*)d_out;
  float* attn = out + (size_t)BB * SS * DD;              // tuple output #2

  const size_t qkvElems = (size_t)BB * HH * SS * DKV;    // 4 Mi elems each
  unsigned short* Qbf = (unsigned short*)d_ws;
  unsigned short* Kbf = Qbf + qkvElems;
  unsigned short* Vbf = Kbf + qkvElems;
  unsigned short* ctx = Vbf + qkvElems;
  float* preLN = (float*)(ctx + (size_t)BB * SS * DD);   // total ~48 MB ws

  const dim3 gg(DD / 64, (BB * SS) / 64);                // (16, 64)
  gemm_xwt_kernel<0,0><<<gg, 256, 0, stream>>>(inQ, Wq, BB * SS, DD, DD, Qbf, nullptr);
  gemm_xwt_kernel<0,0><<<gg, 256, 0, stream>>>(inK, Wk, BB * SS, DD, DD, Kbf, nullptr);
  gemm_xwt_kernel<0,0><<<gg, 256, 0, stream>>>(inV, Wv, BB * SS, DD, DD, Vbf, nullptr);

  const int shbytes = 16 * 2056 * 4;                     // 131584 B (<320KB WGP LDS)
  hipFuncSetAttribute(reinterpret_cast<const void*>(attn_scores_softmax_kernel),
                      hipFuncAttributeMaxDynamicSharedMemorySize, shbytes);
  attn_scores_softmax_kernel<<<dim3(SS / 16, BB * HH), 256, shbytes, stream>>>(
      Qbf, Kbf, mask, attn);

  attn_context_kernel<<<dim3(SS / 16, BB * HH), 128, 0, stream>>>(attn, Vbf, ctx);

  gemm_xwt_kernel<1,1><<<gg, 256, 0, stream>>>(ctx, Wfc, BB * SS, DD, DD, nullptr, preLN);

  ln_residual_kernel<<<BB * SS, 256, 0, stream>>>(preLN, inQ, out);
}
